// LSTM_52329881534788
// MI455X (gfx1250) — matrix operand order repaired and checked by reading, not compiled, not run on hardware
//
#include <hip/hip_runtime.h>
#include <hip/hip_bf16.h>

// ---------------------------------------------------------------------------
// 2-layer LSTM scan (BS=4096, T=31, H=256) for gfx1250 using wave32 WMMA.
//  - pack_b:   fp32 weights -> f16 tiles in the exact V_WMMA B-matrix layout
//  - lstm_scan: persistent blocks; 32 batch rows per block; x_in/h/c in LDS;
//               all GEMMs via v_wmma_f32_16x16x32_f16 (f32 accumulate).
// ---------------------------------------------------------------------------

typedef _Float16 h16;
typedef __attribute__((ext_vector_type(16))) _Float16 v16h;
typedef __attribute__((ext_vector_type(8)))  _Float16 v8h;
typedef __attribute__((ext_vector_type(8)))  float    v8f;

#define BS   4096
#define T    31
#define DIN  124      // layer-0 input width (padded to 128)
#define H    256
#define FEA  64
#define V    20
#define NC   100

#define ROWS 32                // batch rows per workgroup
#define MT   (ROWS / 16)       // 16-row m-tiles per block
#define NBLK (BS / ROWS)
#define NTHREADS 256
#define NWAVES   8

// Packed-B workspace layout (units: halfs).  One tile = 512 halfs:
//   tile[lane*16 + j] = W[nt*16 + (lane&15)][kt*32 + (j&7) + 8*(lane>=16) + 16*(j>=8)]
// which is exactly the 16-bit B-matrix 32x16 VGPR layout for wave32 WMMA.
#define KT_XC  2   // Wxc : N=256 (16 nt), K=64
#define KT_IH0 4   // Wih0: N=1024(64 nt), K=124 -> 128
#define KT_H   8   // Whh0/Wih1/Whh1/Wout/Wcv: K=256
#define OFF_XC   0
#define OFF_IH0  (OFF_XC  + 16 * KT_XC  * 512)
#define OFF_HH0  (OFF_IH0 + 64 * KT_IH0 * 512)
#define OFF_IH1  (OFF_HH0 + 64 * KT_H   * 512)
#define OFF_HH1  (OFF_IH1 + 64 * KT_H   * 512)
#define OFF_OUT  (OFF_HH1 + 64 * KT_H   * 512)   // Wout: N=20->32 (2 nt)
#define OFF_CV   (OFF_OUT + 2  * KT_H   * 512)   // Wcv : N=100->112 (7 nt)
#define WS_HALFS (OFF_CV  + 7  * KT_H   * 512)   // ~1.94 MB of workspace

// LDS layout (bytes)
#define XIN_OFF 0                           // [ROWS][128] f16      (8 KB)
#define H0_OFF  8192                        // 2 x [ROWS][256] f16  (32 KB)
#define H1_OFF  (H0_OFF + 2 * ROWS * 256 * 2)
#define C0_OFF  (H1_OFF + 2 * ROWS * 256 * 2)   // [ROWS][256] f32 (32 KB)
#define C1_OFF  (C0_OFF + ROWS * 256 * 4)
#define OUT_OFF (C1_OFF + ROWS * 256 * 4)       // [ROWS][144] f32 (18 KB)
#define SMEM_BYTES (OUT_OFF + ROWS * 144 * 4)   // 157,696 B < 320 KB/WGP

// ---------------------------------------------------------------------------
__global__ void pack_b(const float* __restrict__ W, h16* __restrict__ dst,
                       int N, int K, int KT, int total) {
  int idx = blockIdx.x * blockDim.x + threadIdx.x;
  if (idx >= total) return;
  int j    = idx & 15;
  int lane = (idx >> 4) & 31;
  int tile = idx >> 9;
  int nt = tile / KT;
  int kt = tile - nt * KT;
  int n = nt * 16 + (lane & 15);
  int k = kt * 32 + (j & 7) + ((lane >> 4) << 3) + ((j >> 3) << 4);
  float v = (n < N && k < K) ? W[n * K + k] : 0.f;
  dst[idx] = (h16)v;
}

// ---------------------------------------------------------------------------
__device__ __forceinline__ float sigmoidf(float x) { return 1.f / (1.f + __expf(-x)); }

__device__ __forceinline__ v8f splat8(float b) {
  v8f r;
#pragma unroll
  for (int i = 0; i < 8; ++i) r[i] = b;
  return r;
}

union AV { v16h v; v8h h[2]; };

// A-matrix 16x32 f16 layout: lane L holds row m*16+(L&15); halfs 0..7 are
// K = kt*32 + 8*(L>=16) + 0..7, halfs 8..15 are that +16.  Both runs are
// 16-byte contiguous in a row-major f16 LDS buffer -> two ds_load_b128.
__device__ __forceinline__ v16h loadA(const h16* base, int strideH, int m,
                                      int kt, int lane) {
  int row = m * 16 + (lane & 15);
  int hi  = lane >> 4;
  const h16* p = base + row * strideH + kt * 32 + hi * 8;
  AV a;
  a.h[0] = *(const v8h*)(p);
  a.h[1] = *(const v8h*)(p + 16);
  return a.v;
}

__device__ __forceinline__ v16h loadB(const h16* tile, int lane) {
  return *(const v16h*)(tile + lane * 16);   // two global_load_b128 (L2-hot)
}

// One LSTM layer for this block's 32-row tile.  Wave w owns m-tile (w>>2)
// and h-tiles [(w&3)*4, +4); it computes all four gate tiles {i,f,g,o} for
// each cell tile so the nonlinearity runs on the WMMA accumulators.
__device__ __forceinline__ void lstm_layer(
    const h16* __restrict__ sAx, int strideX, int ktX, const h16* __restrict__ Pih,
    const h16* __restrict__ sAh, const h16* __restrict__ Phh,
    const float* __restrict__ bih, const float* __restrict__ bhh,
    float* __restrict__ sC, h16* __restrict__ sHout, int lane, int wave) {
  const int m = wave >> 2;
  const int jbase = (wave & 3) * 4;
  for (int jj = 0; jj < 4; ++jj) {
    int j   = jbase + jj;
    int col = j * 16 + (lane & 15);
    v8f acc[4];
#pragma unroll
    for (int g = 0; g < 4; ++g) {
      int gcol = g * H + col;                       // torch gate order i,f,g,o
      acc[g] = splat8(bih[gcol] + bhh[gcol]);
    }
    // x (or h0) contribution
    for (int kt = 0; kt < ktX; ++kt) {
      v16h a = loadA(sAx, strideX, m, kt, lane);
      __builtin_prefetch(Pih + ((0 * 16 + j) * ktX + kt + 1) * 512 + lane * 16, 0, 1);
#pragma unroll
      for (int g = 0; g < 4; ++g) {
        v16h b = loadB(Pih + ((g * 16 + j) * ktX + kt) * 512, lane);
        acc[g] = __builtin_amdgcn_wmma_f32_16x16x32_f16(false, a, false, b,
                                                        (short)0, acc[g], false, false);
      }
    }
    // recurrent contribution
    for (int kt = 0; kt < KT_H; ++kt) {
      v16h a = loadA(sAh, 256, m, kt, lane);
#pragma unroll
      for (int g = 0; g < 4; ++g) {
        v16h b = loadB(Phh + ((g * 16 + j) * KT_H + kt) * 512, lane);
        acc[g] = __builtin_amdgcn_wmma_f32_16x16x32_f16(false, a, false, b,
                                                        (short)0, acc[g], false, false);
      }
    }
    // cell update on accumulator registers (C/D layout: row = m*16+r+8*hi)
    int hi = lane >> 4;
#pragma unroll
    for (int r = 0; r < 8; ++r) {
      int idx = (m * 16 + r + 8 * hi) * 256 + col;
      float c  = sC[idx];
      float ig = sigmoidf(acc[0][r]);
      float fg = sigmoidf(acc[1][r]);
      float gg = tanhf(acc[2][r]);
      float og = sigmoidf(acc[3][r]);
      float cn = fg * c + ig * gg;
      sC[idx]    = cn;
      sHout[idx] = (h16)(og * tanhf(cn));
    }
  }
}

// ---------------------------------------------------------------------------
__global__ __launch_bounds__(NTHREADS, 1) void lstm_scan(
    const float* __restrict__ x, const int* __restrict__ choices,
    const unsigned char* __restrict__ masks,    // bool tensors arrive as bytes
    const unsigned char* __restrict__ cidx, const int* __restrict__ cchoice,
    const float* __restrict__ bih0, const float* __restrict__ bhh0,
    const float* __restrict__ bih1, const float* __restrict__ bhh1,
    const float* __restrict__ bout, const float* __restrict__ bxc,
    const float* __restrict__ bcv, const h16* __restrict__ P,
    float* __restrict__ out) {
  extern __shared__ char smem[];
  h16*   sXin = (h16*)(smem + XIN_OFF);   // [ROWS][128], padded, zero-init
  h16*   sH0  = (h16*)(smem + H0_OFF);    // double-buffered
  h16*   sH1  = (h16*)(smem + H1_OFF);
  float* sC0  = (float*)(smem + C0_OFF);
  float* sC1  = (float*)(smem + C1_OFF);
  float* sOut = (float*)(smem + OUT_OFF); // [ROWS][144]: V pad 32 | NC pad 112
  h16*   sXf  = (h16*)sOut;               // alias: f16 staging of x (init only)

  const int tid  = threadIdx.x;
  const int lane = tid & 31;
  const int wave = tid >> 5;
  const int bb   = blockIdx.x * ROWS;

  for (int i = tid; i < ROWS * 128; i += NTHREADS) sXin[i] = (h16)0.f;
  for (int i = tid; i < ROWS * 256; i += NTHREADS) { sH0[i] = (h16)0.f; sH1[i] = (h16)0.f; }
  for (int i = tid; i < ROWS * FEA; i += NTHREADS)
    sXf[i] = (h16)x[(size_t)(bb + (i >> 6)) * FEA + (i & 63)];
  __syncthreads();

  // c0 = x @ Wxc^T + bxc, replicated into both layers' cell state.
  {
    const int m = wave >> 2;
    const int jbase = (wave & 3) * 4;
    for (int jj = 0; jj < 4; ++jj) {
      int j   = jbase + jj;
      int col = j * 16 + (lane & 15);
      v8f acc = splat8(bxc[col]);
      for (int kt = 0; kt < KT_XC; ++kt) {
        v16h a = loadA(sXf, FEA, m, kt, lane);
        v16h b = loadB(P + OFF_XC + (j * KT_XC + kt) * 512, lane);
        acc = __builtin_amdgcn_wmma_f32_16x16x32_f16(false, a, false, b,
                                                     (short)0, acc, false, false);
      }
      int hi = lane >> 4;
#pragma unroll
      for (int r = 0; r < 8; ++r) {
        int idx = (m * 16 + r + 8 * hi) * 256 + col;
        sC0[idx] = acc[r];
        sC1[idx] = acc[r];
      }
    }
  }
  __syncthreads();

  float lp = 0.f;
  int cur = 0;
  for (int t = 0; t < T; ++t) {
    const h16* h0c = sH0 + cur * ROWS * 256;
    h16*       h0n = sH0 + (1 - cur) * ROWS * 256;
    const h16* h1c = sH1 + cur * ROWS * 256;
    h16*       h1n = sH1 + (1 - cur) * ROWS * 256;

    lstm_layer(sXin, 128, KT_IH0, P + OFF_IH0, h0c, P + OFF_HH0,
               bih0, bhh0, sC0, h0n, lane, wave);
    __syncthreads();   // all layer-0 reads of sXin / h0c complete

    // scatter this step's 4-bit token code (consumed from t+1 onward)
    if (tid < ROWS * 4) {
      int row = tid >> 2, b = tid & 3;
      int ch = choices[(size_t)(bb + row) * T + t];
      sXin[row * 128 + t * 4 + b] = (h16)((ch >> (3 - b)) & 1);
    }

    lstm_layer(h0n, 256, KT_H, P + OFF_IH1, h1c, P + OFF_HH1,
               bih1, bhh1, sC1, h1n, lane, wave);
    __syncthreads();   // h1n fully written before the heads read it

    // output heads: logits = h1 @ Wout^T + bout, clog = h1 @ Wcv^T + bcv
    for (int tt = wave; tt < MT * 9; tt += NWAVES) {
      int m = tt / 9, nIdx = tt % 9;
      int cl = lane & 15;
      const h16* Pb;
      int outCol;
      float bias;
      if (nIdx < 2) {
        int col = nIdx * 16 + cl;
        Pb = P + OFF_OUT + (size_t)nIdx * KT_H * 512;
        outCol = col;
        bias = (col < V) ? bout[col] : 0.f;
      } else {
        int col = (nIdx - 2) * 16 + cl;
        Pb = P + OFF_CV + (size_t)(nIdx - 2) * KT_H * 512;
        outCol = 32 + col;
        bias = (col < NC) ? bcv[col] : 0.f;
      }
      v8f acc = splat8(bias);
      for (int kt = 0; kt < KT_H; ++kt) {
        v16h a = loadA(h1n, 256, m, kt, lane);
        v16h b = loadB(Pb + kt * 512, lane);
        acc = __builtin_amdgcn_wmma_f32_16x16x32_f16(false, a, false, b,
                                                     (short)0, acc, false, false);
      }
      int hi = lane >> 4;
#pragma unroll
      for (int r = 0; r < 8; ++r)
        sOut[(m * 16 + r + 8 * hi) * 144 + outCol] = acc[r];
    }
    __syncthreads();

    // masked log-softmax + gathers, one row per lane
    if (tid < ROWS) {
      size_t gb = (size_t)(bb + tid);
      const float* Lrow = sOut + tid * 144;
      const unsigned char* mrow = masks + (gb * T + t) * V;
      float mx = -1e30f;
      for (int v = 0; v < V; ++v) { float z = mrow[v] ? Lrow[v] : -1e9f; mx = fmaxf(mx, z); }
      float se = 0.f;
      for (int v = 0; v < V; ++v) { float z = mrow[v] ? Lrow[v] : -1e9f; se += __expf(z - mx); }
      float lse = mx + __logf(se);
      int ch = choices[gb * T + t];
      lp += (mrow[ch] ? Lrow[ch] : -1e9f) - lse;

      const float* Crow = Lrow + 32;
      float mx2 = -1e30f;
      for (int c = 0; c < NC; ++c) mx2 = fmaxf(mx2, Crow[c]);
      float se2 = 0.f;
      for (int c = 0; c < NC; ++c) se2 += __expf(Crow[c] - mx2);
      float lse2 = mx2 + __logf(se2);
      if (cidx[gb * T + t]) lp += Crow[cchoice[gb * T + t]] - lse2;
    }
    __syncthreads();
    cur ^= 1;
  }
  if (tid < ROWS) out[bb + tid] = lp;
}

// ---------------------------------------------------------------------------
extern "C" void kernel_launch(void* const* d_in, const int* in_sizes, int n_in,
                              void* d_out, int out_size, void* d_ws, size_t ws_size,
                              hipStream_t stream) {
  (void)in_sizes; (void)n_in; (void)out_size; (void)ws_size;
  const float* x       = (const float*)d_in[0];
  const int*   choices = (const int*)d_in[1];
  const unsigned char* masks = (const unsigned char*)d_in[2];
  const unsigned char* cidx  = (const unsigned char*)d_in[3];
  const int*   cchoice = (const int*)d_in[4];
  const float* Wih0 = (const float*)d_in[5];
  const float* Whh0 = (const float*)d_in[6];
  const float* bih0 = (const float*)d_in[7];
  const float* bhh0 = (const float*)d_in[8];
  const float* Wih1 = (const float*)d_in[9];
  const float* Whh1 = (const float*)d_in[10];
  const float* bih1 = (const float*)d_in[11];
  const float* bhh1 = (const float*)d_in[12];
  const float* Wout = (const float*)d_in[13];
  const float* bout = (const float*)d_in[14];
  const float* Wxc  = (const float*)d_in[15];
  const float* bxc  = (const float*)d_in[16];
  const float* Wcv  = (const float*)d_in[17];
  const float* bcv  = (const float*)d_in[18];
  h16* P = (h16*)d_ws;   // ~1.94 MB of packed f16 B-tiles

  auto pack = [&](const float* W, size_t off, int N, int K, int KT) {
    int total = ((N + 15) / 16) * KT * 512;
    pack_b<<<(total + 255) / 256, 256, 0, stream>>>(W, P + off, N, K, KT, total);
  };
  pack(Wxc,  OFF_XC,  256,  64,  KT_XC);
  pack(Wih0, OFF_IH0, 1024, 124, KT_IH0);
  pack(Whh0, OFF_HH0, 1024, 256, KT_H);
  pack(Wih1, OFF_IH1, 1024, 256, KT_H);
  pack(Whh1, OFF_HH1, 1024, 256, KT_H);
  pack(Wout, OFF_OUT, 20,   256, KT_H);
  pack(Wcv,  OFF_CV,  100,  256, KT_H);

  (void)hipFuncSetAttribute((const void*)lstm_scan,
                            hipFuncAttributeMaxDynamicSharedMemorySize,
                            (int)SMEM_BYTES);
  lstm_scan<<<NBLK, NTHREADS, SMEM_BYTES, stream>>>(
      x, choices, masks, cidx, cchoice, bih0, bhh0, bih1, bhh1,
      bout, bxc, bcv, P, (float*)d_out);
}
